// CausalSelfAttention_78529182040824
// MI455X (gfx1250) — matrix operand order repaired and checked by reading, not compiled, not run on hardware
//
#include <hip/hip_runtime.h>
#include <hip/hip_bf16.h>

#define BATCH 2
#define SEQ   2048
#define DMODEL 1024
#define NHEAD 16
#define HDIM  64

typedef __attribute__((ext_vector_type(16))) __bf16 bf16x16;
typedef __attribute__((ext_vector_type(8)))  __bf16 bf16x8;
typedef __attribute__((ext_vector_type(8)))  float  v8f;

// float -> bf16 with round-to-nearest-even, via bit ops
static __device__ inline __bf16 f2bf(float f) {
    union { float f; unsigned u; } a; a.f = f;
    unsigned u = a.u;
    u += 0x7fffu + ((u >> 16) & 1u);
    union { unsigned short s; __bf16 b; } o; o.s = (unsigned short)(u >> 16);
    return o.b;
}

// load a 16-element bf16 fragment as two 16B chunks (global or LDS)
static __device__ inline bf16x16 load2(const __bf16* p0, const __bf16* p1) {
    union { bf16x16 v; uint4 q[2]; } f;
    f.q[0] = *(const uint4*)p0;
    f.q[1] = *(const uint4*)p1;
    return f.v;
}

static __device__ inline v8f wmma_bf16(bf16x16 a, bf16x16 b, v8f c) {
    return __builtin_amdgcn_wmma_f32_16x16x32_bf16(false, a, false, b, (short)0, c, false, false);
}

// async 16B copy global -> LDS (gfx1250, tracked by ASYNCcnt)
static __device__ inline void async_g2l_b128(const __bf16* gsrc, __bf16* ldst) {
    unsigned loff = (unsigned)(size_t)ldst;   // generic low 32 bits == LDS offset
    asm volatile("global_load_async_to_lds_b128 %0, %1, off"
                 :: "v"(loff), "v"(gsrc) : "memory");
}
static __device__ inline void wait_async0() {
    asm volatile("s_wait_asynccnt 0x0" ::: "memory");
}

// ---------------- kernel 1: x (f32) -> xb (bf16), 8 elems/thread ----------------
__global__ void k_cvt_x(const float* __restrict__ x, __bf16* __restrict__ xb) {
    int i = (blockIdx.x * blockDim.x + threadIdx.x) * 8;
    float4 a = *(const float4*)(x + i);
    float4 b = *(const float4*)(x + i + 4);
    union { bf16x8 v; uint4 u; } o;
    o.v[0] = f2bf(a.x); o.v[1] = f2bf(a.y); o.v[2] = f2bf(a.z); o.v[3] = f2bf(a.w);
    o.v[4] = f2bf(b.x); o.v[5] = f2bf(b.y); o.v[6] = f2bf(b.z); o.v[7] = f2bf(b.w);
    *(uint4*)(xb + i) = o.u;
}

// ---------------- kernel 2: pack W into per-lane B-fragment order ----------------
// tile t = kt*64 + nt; lane l (n=l&15, h=l>>4), elem j: W[kt*32 + h*16 + j][nt*16 + n]
__global__ void k_pack_w(const float* __restrict__ wq, const float* __restrict__ wk,
                         const float* __restrict__ wv, __bf16* __restrict__ wpack) {
    int tid  = blockIdx.x * blockDim.x + threadIdx.x;   // 0 .. 3*2048*32-1
    int widx = tid / (2048 * 32);
    int rem  = tid % (2048 * 32);
    int t = rem >> 5, l = rem & 31;
    int kt = t >> 6, nt = t & 63;
    int n = l & 15, h = l >> 4;
    const float* w = (widx == 0) ? wq : ((widx == 1) ? wk : wv);
    int col  = nt * 16 + n;
    int row0 = kt * 32 + h * 16;
    union { bf16x16 v; uint4 q[2]; } o;
#pragma unroll
    for (int j = 0; j < 16; ++j)
        o.v[j] = f2bf(w[(size_t)(row0 + j) * DMODEL + col]);
    __bf16* dst = wpack + (size_t)tid * 16;
    *(uint4*)dst       = o.q[0];
    *((uint4*)dst + 1) = o.q[1];
}

// ---------------- kernel 3: QKV projection GEMM (one wave = 16x64 head tile) -----
// grid: (SEQ/16, NHEAD, BATCH*3), block: 32. A fragment reused across 4 N-tiles.
__global__ void k_proj(const __bf16* __restrict__ xb, const __bf16* __restrict__ wpack,
                       const float* __restrict__ bq, const float* __restrict__ bk,
                       const float* __restrict__ bv,
                       __bf16* __restrict__ qb, __bf16* __restrict__ kb,
                       __bf16* __restrict__ vt) {
    int lane = threadIdx.x;
    int s0   = blockIdx.x * 16;
    int head = blockIdx.y;
    int z    = blockIdx.z;
    int b = z / 3, which = z % 3;
    int n = lane & 15, h = lane >> 4;

    const float* bias = (which == 0) ? bq : ((which == 1) ? bk : bv);
    v8f acc[4];
#pragma unroll
    for (int t = 0; t < 4; ++t) {
        float bval = bias[head * HDIM + t * 16 + n];
#pragma unroll
        for (int r = 0; r < 8; ++r) acc[t][r] = bval;
    }

    const __bf16* xrow  = xb + ((size_t)(b * SEQ + s0 + n)) * DMODEL;  // A-row = lane&15
    const __bf16* wbase = wpack + (size_t)which * 2048 * 32 * 16;

    for (int kt = 0; kt < 32; ++kt) {
        bf16x16 a = load2(xrow + kt * 32 + h * 16, xrow + kt * 32 + h * 16 + 8);
#pragma unroll
        for (int t = 0; t < 4; ++t) {
            int nt = head * 4 + t;
            const __bf16* wp = wbase + ((size_t)(kt * 64 + nt) * 32 + lane) * 16;
            bf16x16 bbf = load2(wp, wp + 8);
            acc[t] = wmma_bf16(a, bbf, acc[t]);
        }
    }

    int bh = b * NHEAD + head;
    if (which < 2) {
        __bf16* out = (which == 0) ? qb : kb;  // row-major [B,H,S,64]
#pragma unroll
        for (int t = 0; t < 4; ++t)
#pragma unroll
            for (int r = 0; r < 8; ++r) {
                int m = r + 8 * h;
                out[((size_t)bh * SEQ + s0 + m) * HDIM + t * 16 + n] = f2bf(acc[t][r]);
            }
    } else {
        // V transposed: [B,H,64,S]; rows m = r+8h contiguous -> one b128 store per tile
#pragma unroll
        for (int t = 0; t < 4; ++t) {
            union { bf16x8 v; uint4 u; } o;
#pragma unroll
            for (int r = 0; r < 8; ++r) o.v[r] = f2bf(acc[t][r]);
            __bf16* dst = vt + ((size_t)bh * HDIM + t * 16 + n) * SEQ + s0 + 8 * h;
            *(uint4*)dst = o.u;
        }
    }
}

// ---------------- kernel 4: flash attention with async K/V staging ---------------
// grid: (SEQ/16/4, NHEAD, BATCH), block: 128 (4 waves, one 16-row q tile each).
// K/V tiles for the shared (b,h) are staged into LDS once per block via
// global_load_async_to_lds_b128, double-buffered across the 32-key loop.
#define KPAD 72   // 32 x 72 bf16 rows (144B stride -> conflict-free b128 reads)
#define VPAD 40   // 64 x 40 bf16 rows (80B stride)
__global__ void k_attn(const __bf16* __restrict__ qb, const __bf16* __restrict__ kb,
                       const __bf16* __restrict__ vt, float* __restrict__ out) {
    __shared__ __bf16 Kbuf[2][32][KPAD];   // 9216 B
    __shared__ __bf16 Vbuf[2][64][VPAD];   // 10240 B
    __shared__ __bf16 Pbuf[4][16][40];     // 5120 B
    int tid  = threadIdx.x;
    int lane = tid & 31;
    int w    = tid >> 5;
    int qt   = blockIdx.x * 4 + w;
    int head = blockIdx.y, b = blockIdx.z;
    int bh = b * NHEAD + head;
    int n = lane & 15, h = lane >> 4;
    int qs0 = qt * 16;

    const __bf16* kbase = kb + (size_t)bh * SEQ * HDIM;   // [key][64]
    const __bf16* vbase = vt + (size_t)bh * HDIM * SEQ;   // [d][key]

    const __bf16* qrow = qb + ((size_t)bh * SEQ + qs0 + n) * HDIM;  // A-row = lane&15
    bf16x16 aQ0 = load2(qrow + h * 16,      qrow + h * 16 + 8);      // d 0..31
    bf16x16 aQ1 = load2(qrow + 32 + h * 16, qrow + 32 + h * 16 + 8); // d 32..63

    v8f y[4];
#pragma unroll
    for (int t = 0; t < 4; ++t) y[t] = (v8f){};
    float mrow[8], lrow[8];
#pragma unroll
    for (int r = 0; r < 8; ++r) { mrow[r] = -1e30f; lrow[r] = 0.f; }

    const float sc = 0.125f;  // 1/sqrt(64)

    // stage K/V tile for key block `key0` into buffer `buf`
    auto stage = [&](int buf, int key0) {
#pragma unroll
        for (int half = 0; half < 2; ++half) {
            int c = tid + half * 128;              // 256 chunks of 16B each
            int key = c >> 3, kc = c & 7;          // K: 32 keys x 8 chunks (64 d)
            async_g2l_b128(kbase + (size_t)(key0 + key) * HDIM + kc * 8,
                           &Kbuf[buf][key][kc * 8]);
            int d = c >> 2, dc = c & 3;            // V: 64 d x 4 chunks (32 keys)
            async_g2l_b128(vbase + (size_t)d * SEQ + key0 + dc * 8,
                           &Vbuf[buf][d][dc * 8]);
        }
    };

    stage(0, 0);
    for (int it = 0; it < SEQ / 32; ++it) {
        int cur = it & 1;
        int key0 = it * 32;
        wait_async0();        // this wave's pending async stores to LDS done
        __syncthreads();      // ...and everyone else's; prev reads of other buf done
        if (it + 1 < SEQ / 32) stage(cur ^ 1, key0 + 32);

        // scores: two 16-key subtiles, K-reduction over d=64
        v8f s[2];
#pragma unroll
        for (int sub = 0; sub < 2; ++sub) {
            const __bf16* krow = &Kbuf[cur][sub * 16 + n][0];
            bf16x16 bk0 = load2(krow + h * 16,      krow + h * 16 + 8);       // d 0..31
            bf16x16 bk1 = load2(krow + 32 + h * 16, krow + 32 + h * 16 + 8);  // d 32..63
            v8f zz = (v8f){};
            zz = wmma_bf16(aQ0, bk0, zz);
            zz = wmma_bf16(aQ1, bk1, zz);
            s[sub] = zz;
        }
        // online softmax; row stats live in 16-lane half-groups (C-layout rows r+8h)
        float p0[8], p1[8];
#pragma unroll
        for (int r = 0; r < 8; ++r) {
            float v0 = s[0][r] * sc, v1 = s[1][r] * sc;
            float tv = fmaxf(v0, v1);
            tv = fmaxf(tv, __shfl_xor(tv, 1, 16));
            tv = fmaxf(tv, __shfl_xor(tv, 2, 16));
            tv = fmaxf(tv, __shfl_xor(tv, 4, 16));
            tv = fmaxf(tv, __shfl_xor(tv, 8, 16));
            float mn = fmaxf(mrow[r], tv);
            float alpha = __expf(mrow[r] - mn);
            mrow[r] = mn;
            float a0 = __expf(v0 - mn);
            float a1 = __expf(v1 - mn);
            float ps = a0 + a1;
            ps += __shfl_xor(ps, 1, 16);
            ps += __shfl_xor(ps, 2, 16);
            ps += __shfl_xor(ps, 4, 16);
            ps += __shfl_xor(ps, 8, 16);
            lrow[r] = lrow[r] * alpha + ps;
#pragma unroll
            for (int t = 0; t < 4; ++t) y[t][r] *= alpha;
            p0[r] = a0; p1[r] = a1;
        }
        // stage P (16x32) through LDS: C-fragment -> A-fragment relayout
#pragma unroll
        for (int r = 0; r < 8; ++r) {
            int m = r + 8 * h;
            Pbuf[w][m][n]      = f2bf(p0[r]);
            Pbuf[w][m][n + 16] = f2bf(p1[r]);
        }
        const __bf16* prow = &Pbuf[w][n][0];               // A-row = lane&15
        bf16x16 aP = load2(prow + h * 16, prow + h * 16 + 8);
        // y += P @ V  (V fragments from the staged transposed tile)
#pragma unroll
        for (int t = 0; t < 4; ++t) {
            const __bf16* vrow = &Vbuf[cur][t * 16 + n][h * 16];
            bf16x16 bvf = load2(vrow, vrow + 8);
            y[t] = wmma_bf16(aP, bvf, y[t]);
        }
    }

    // finalize: divide by row sums, write f32 output [B,S,16*64]
#pragma unroll
    for (int r = 0; r < 8; ++r) {
        int m = r + 8 * h;
        float inv = 1.f / lrow[r];
        float* orow = out + ((size_t)(b * SEQ + qs0 + m)) * DMODEL + head * HDIM;
#pragma unroll
        for (int t = 0; t < 4; ++t) orow[t * 16 + n] = y[t][r] * inv;
    }
}

extern "C" void kernel_launch(void* const* d_in, const int* in_sizes, int n_in,
                              void* d_out, int out_size, void* d_ws, size_t ws_size,
                              hipStream_t stream) {
    (void)in_sizes; (void)n_in; (void)out_size; (void)ws_size;
    const float* x  = (const float*)d_in[0];
    const float* wq = (const float*)d_in[1];
    const float* bq = (const float*)d_in[2];
    const float* wk = (const float*)d_in[3];
    const float* bk = (const float*)d_in[4];
    const float* wv = (const float*)d_in[5];
    const float* bv = (const float*)d_in[6];

    char* ws = (char*)d_ws;
    __bf16* xb    = (__bf16*)(ws);                              //  8 MB
    __bf16* wpack = (__bf16*)(ws + (size_t)8  * 1024 * 1024);   //  6 MB
    __bf16* qb    = (__bf16*)(ws + (size_t)14 * 1024 * 1024);   //  8 MB
    __bf16* kb    = (__bf16*)(ws + (size_t)22 * 1024 * 1024);   //  8 MB
    __bf16* vt    = (__bf16*)(ws + (size_t)30 * 1024 * 1024);   //  8 MB (total 38 MB)

    int nx = BATCH * SEQ * DMODEL;               // 4,194,304
    k_cvt_x<<<nx / (256 * 8), 256, 0, stream>>>(x, xb);
    k_pack_w<<<(3 * 2048 * 32) / 256, 256, 0, stream>>>(wq, wk, wv, wpack);
    dim3 gp(SEQ / 16, NHEAD, BATCH * 3);
    k_proj<<<gp, 32, 0, stream>>>(xb, wpack, bq, bk, bv, qb, kb, vt);
    dim3 ga(SEQ / 16 / 4, NHEAD, BATCH);
    k_attn<<<ga, 128, 0, stream>>>(qb, kb, vt, (float*)d_out);
}
